// ShotGAT_56530359549967
// MI455X (gfx1250) — matrix operand order repaired and checked by reading, not compiled, not run on hardware
//
#include <hip/hip_runtime.h>
#include <math.h>

#define NN 10000
#define EE 160000
#define HH 8
#define CC 128
#define DD 1024
#define GG 100
#define NFD 15
#define EFD 7
#define NP 10048   // NN padded to a multiple of 64 (GEMM M-strip size)

typedef __attribute__((ext_vector_type(16))) __bf16 v16bf;
typedef __attribute__((ext_vector_type(8)))  float  v8f;

struct U32x8 { uint4 a, b; };   // 32-byte bag -> bit_cast to v16bf

// ---------- helpers ----------
__device__ __forceinline__ unsigned short f2bf(float f) {
  unsigned u = __builtin_bit_cast(unsigned, f);
  u += 0x7FFFu + ((u >> 16) & 1u);              // round-to-nearest-even
  return (unsigned short)(u >> 16);
}
// order-preserving float <-> uint mapping (for atomic max on floats)
__device__ __forceinline__ unsigned fmap(float f) {
  unsigned u = __builtin_bit_cast(unsigned, f);
  return (u & 0x80000000u) ? ~u : (u | 0x80000000u);
}
__device__ __forceinline__ float funmap(unsigned u) {
  unsigned v = (u & 0x80000000u) ? (u & 0x7FFFFFFFu) : ~u;
  return __builtin_bit_cast(float, v);
}
#define FMAP_NEG_INF 0x007FFFFFu   // fmap(-inf)

__device__ __forceinline__ v16bf load_frag(const unsigned short* p) {
  U32x8 u;
  u.a = *(const uint4*)(p);
  u.b = *(const uint4*)(p + 8);
  return __builtin_bit_cast(v16bf, u);
}

// ---------- utility kernels ----------
__global__ void fill_f32(float* p, float v, int n) {
  int i = blockIdx.x * blockDim.x + threadIdx.x;
  if (i < n) p[i] = v;
}
__global__ void fill_u32(unsigned* p, unsigned v, int n) {
  int i = blockIdx.x * blockDim.x + threadIdx.x;
  if (i < n) p[i] = v;
}
__global__ void cvt_bf16(unsigned short* dst, const float* __restrict__ src, int n) {
  int i = blockIdx.x * blockDim.x + threadIdx.x;
  if (i < n) dst[i] = f2bf(src[i]);
}

// pack W (DD x DD f32, row-major) into WMMA-B register layout (bf16):
//   Bp[((t*DD + n)*2 + hh)*16 + i] = bf16( W[(t*32 + hh*16 + i)*DD + n] )
// so a lane (column n, K-half hh) reads its whole fragment as 2x b128.
__global__ void pack_b_bf16(unsigned short* __restrict__ Bp, const float* __restrict__ W) {
  int idx = blockIdx.x * blockDim.x + threadIdx.x;
  if (idx >= DD * DD) return;
  int i  = idx & 15;
  int hh = (idx >> 4) & 1;
  int n  = (idx >> 5) & (DD - 1);
  int t  = idx >> 15;                 // / (DD*32)
  int k  = t * 32 + hh * 16 + i;
  Bp[idx] = f2bf(W[k * DD + n]);
}

// ---------- small fp32 GEMM (layer-1 transforms, K=15) ----------
__global__ void gemm_f32_small(const float* __restrict__ A, const float* __restrict__ W,
                               float* __restrict__ C, int M, int K, int Nc) {
  int i = blockIdx.x * blockDim.x + threadIdx.x;
  if (i >= M * Nc) return;
  int n = i / Nc, j = i - n * Nc;
  float s = 0.f;
  for (int k = 0; k < K; ++k) s += A[n * K + k] * W[k * Nc + j];
  C[i] = s;
}

// ---------- bf16 WMMA GEMM: C[NP x DD] = A[NP x DD] * B[DD x DD] ----------
// A row-major bf16 (NP rows, zero-padded past NN); Bp pre-packed (pack_b_bf16).
// One wave computes a 64x16 strip of C (4 M-tiles reuse one B fragment).
// Software-pipelined: iteration t+1's 10 b128 loads are issued before
// iteration t's WMMAs so the backend can emit partial loadcnt waits.
// Grid is exact: no guards, uniform flow, EXEC all-ones throughout.
__global__ __launch_bounds__(256) void gemm_bf16_wmma(
    const unsigned short* __restrict__ A, const unsigned short* __restrict__ Bp,
    float* __restrict__ C) {
  int wid   = (blockIdx.x * blockDim.x + threadIdx.x) >> 5;
  int lane  = threadIdx.x & 31;
  int strip = wid >> 6;                 // 64 N-tiles per strip row
  int tn    = wid & 63;
  int half  = lane >> 4;                // K-half: 0 or 1
  int mr    = lane & 15;                // row (A) / column (B,C) within tile
  int r0    = strip * 64;
  const unsigned short* Bl = Bp + (((size_t)(tn * 16 + mr) * 2 + half) << 4);
  const unsigned short* Ap = A + (size_t)(r0 + mr) * DD + half * 16;
  v8f acc0 = (v8f){}, acc1 = (v8f){}, acc2 = (v8f){}, acc3 = (v8f){};
  // prologue: fragments for t = 0
  v16bf bc  = load_frag(Bl);
  v16bf a0c = load_frag(Ap);
  v16bf a1c = load_frag(Ap + 16 * DD);
  v16bf a2c = load_frag(Ap + 32 * DD);
  v16bf a3c = load_frag(Ap + 48 * DD);
  for (int t = 0; t < DD / 32 - 1; ++t) {
    Bl += (size_t)DD * 32;
    Ap += 32;
    // prefetch t+1 (stays in flight across the WMMAs below)
    v16bf bn  = load_frag(Bl);
    v16bf a0n = load_frag(Ap);
    v16bf a1n = load_frag(Ap + 16 * DD);
    v16bf a2n = load_frag(Ap + 32 * DD);
    v16bf a3n = load_frag(Ap + 48 * DD);
    acc0 = __builtin_amdgcn_wmma_f32_16x16x32_bf16(false, a0c, false, bc, (short)0, acc0, false, false);
    acc1 = __builtin_amdgcn_wmma_f32_16x16x32_bf16(false, a1c, false, bc, (short)0, acc1, false, false);
    acc2 = __builtin_amdgcn_wmma_f32_16x16x32_bf16(false, a2c, false, bc, (short)0, acc2, false, false);
    acc3 = __builtin_amdgcn_wmma_f32_16x16x32_bf16(false, a3c, false, bc, (short)0, acc3, false, false);
    bc = bn; a0c = a0n; a1c = a1n; a2c = a2n; a3c = a3n;
  }
  // epilogue: last k-step
  acc0 = __builtin_amdgcn_wmma_f32_16x16x32_bf16(false, a0c, false, bc, (short)0, acc0, false, false);
  acc1 = __builtin_amdgcn_wmma_f32_16x16x32_bf16(false, a1c, false, bc, (short)0, acc1, false, false);
  acc2 = __builtin_amdgcn_wmma_f32_16x16x32_bf16(false, a2c, false, bc, (short)0, acc2, false, false);
  acc3 = __builtin_amdgcn_wmma_f32_16x16x32_bf16(false, a3c, false, bc, (short)0, acc3, false, false);
  // C/D f32 16x16: VGPR j -> row j + 8*half, col = mr
  float* Cp = C + (size_t)(r0 + half * 8) * DD + tn * 16 + mr;
#pragma unroll
  for (int j = 0; j < 8; ++j) Cp[(size_t)j * DD] = acc0[j];
  Cp += (size_t)16 * DD;
#pragma unroll
  for (int j = 0; j < 8; ++j) Cp[(size_t)j * DD] = acc1[j];
  Cp += (size_t)16 * DD;
#pragma unroll
  for (int j = 0; j < 8; ++j) Cp[(size_t)j * DD] = acc2[j];
  Cp += (size_t)16 * DD;
#pragma unroll
  for (int j = 0; j < 8; ++j) Cp[(size_t)j * DD] = acc3[j];
}

// ---------- edge attention: one wave per (edge, head) ----------
__global__ __launch_bounds__(256) void edge_logit(
    const float* __restrict__ xl, const float* __restrict__ xr,
    const float* __restrict__ eattr, const float* __restrict__ We,
    const float* __restrict__ att, const int* __restrict__ src,
    const int* __restrict__ dst, float* __restrict__ logit,
    unsigned* __restrict__ nmax) {
  int wid = (blockIdx.x * blockDim.x + threadIdx.x) >> 5;
  if (wid >= EE * HH) return;
  int e = wid >> 3, h = wid & 7;
  int lane = threadIdx.x & 31;
  int s = src[e], d = dst[e];
  int cbase = h * CC + lane * 4;
  // edge transform on the fly: ea = edge_attr[e] @ We  (EFD=7 FMAs per channel)
  float ea0 = 0.f, ea1 = 0.f, ea2 = 0.f, ea3 = 0.f;
#pragma unroll
  for (int k = 0; k < EFD; ++k) {
    float ek = eattr[e * EFD + k];
    float4 wv = *(const float4*)(We + k * DD + cbase);
    ea0 += ek * wv.x; ea1 += ek * wv.y; ea2 += ek * wv.z; ea3 += ek * wv.w;
  }
  float4 lv = *(const float4*)(xl + (size_t)s * DD + cbase);
  float4 rv = *(const float4*)(xr + (size_t)d * DD + cbase);
  float4 av = *(const float4*)(att + cbase);
  float m0 = lv.x + rv.x + ea0; m0 = m0 > 0.f ? m0 : 0.2f * m0;
  float m1 = lv.y + rv.y + ea1; m1 = m1 > 0.f ? m1 : 0.2f * m1;
  float m2 = lv.z + rv.z + ea2; m2 = m2 > 0.f ? m2 : 0.2f * m2;
  float m3 = lv.w + rv.w + ea3; m3 = m3 > 0.f ? m3 : 0.2f * m3;
  float sum = m0 * av.x + m1 * av.y + m2 * av.z + m3 * av.w;
#pragma unroll
  for (int off = 16; off > 0; off >>= 1) sum += __shfl_down(sum, off, 32);
  if (lane == 0) {
    logit[wid] = sum;
    atomicMax(&nmax[d * HH + h], fmap(sum));
  }
}

__global__ void edge_exp(float* __restrict__ logit, const unsigned* __restrict__ nmax,
                         float* __restrict__ den, const int* __restrict__ dst) {
  int i = blockIdx.x * blockDim.x + threadIdx.x;
  if (i >= EE * HH) return;
  int e = i >> 3, h = i & 7;
  int d = dst[e];
  float mx = funmap(nmax[d * HH + h]);
  if (!(mx >= -3.0e38f)) mx = 0.f;               // -inf (no-edge node) -> 0, like reference
  float ex = expf(logit[i] - mx);
  logit[i] = ex;                                  // reuse buffer for ex
  atomicAdd(&den[d * HH + h], ex);
}

__global__ __launch_bounds__(256) void edge_aggr(
    const float* __restrict__ xl, const float* __restrict__ exv,
    const float* __restrict__ den, const int* __restrict__ src,
    const int* __restrict__ dst, float* __restrict__ outb) {
  int wid = (blockIdx.x * blockDim.x + threadIdx.x) >> 5;
  if (wid >= EE * HH) return;
  int e = wid >> 3, h = wid & 7;
  int lane = threadIdx.x & 31;
  int s = src[e], d = dst[e];
  float alpha = exv[wid] / (den[d * HH + h] + 1e-16f);
  int cbase = h * CC + lane * 4;
  float4 lv = *(const float4*)(xl + (size_t)s * DD + cbase);
  float* op = outb + (size_t)d * DD + cbase;
  atomicAdd(op + 0, lv.x * alpha);
  atomicAdd(op + 1, lv.y * alpha);
  atomicAdd(op + 2, lv.z * alpha);
  atomicAdd(op + 3, lv.w * alpha);
}

// ---------- bias + BN(eval) + ELU ----------
__global__ void post_bn_elu(const float* __restrict__ outb, const float* __restrict__ bias,
                            const float* __restrict__ g, const float* __restrict__ b,
                            float* __restrict__ h) {
  int i = blockIdx.x * blockDim.x + threadIdx.x;
  if (i >= NN * DD) return;
  int dd = i & (DD - 1);
  float v = outb[i] + bias[dd];
  v = v * (g[dd] * rsqrtf(1.f + 1e-5f)) + b[dd];
  h[i] = v > 0.f ? v : expm1f(v);
}

// ---------- pooling ----------
__global__ void pool_count(const int* __restrict__ batch, float* __restrict__ counts) {
  int n = blockIdx.x * blockDim.x + threadIdx.x;
  if (n < NN) atomicAdd(&counts[batch[n]], 1.f);
}
__global__ void pool_sum(const float* __restrict__ h, const int* __restrict__ batch,
                         float* __restrict__ pooled) {
  int i = blockIdx.x * blockDim.x + threadIdx.x;
  if (i >= NN * DD) return;
  int n = i >> 10, dd = i & 1023;
  atomicAdd(&pooled[batch[n] * DD + dd], h[i]);
}
__global__ void pool_div(float* __restrict__ pooled, const float* __restrict__ counts) {
  int i = blockIdx.x * blockDim.x + threadIdx.x;
  if (i >= GG * DD) return;
  pooled[i] /= fmaxf(counts[i >> 10], 1.f);
}

// ---------- MLP head ----------
__global__ void fc_elu(const float* __restrict__ in, const float* __restrict__ w,
                       const float* __restrict__ bias, float* __restrict__ out,
                       int M, int K, int Nc) {
  int i = blockIdx.x * blockDim.x + threadIdx.x;
  if (i >= M * Nc) return;
  int g = i / Nc, j = i - g * Nc;
  float s = bias[j];
  for (int k = 0; k < K; ++k) s += in[g * K + k] * w[k * Nc + j];
  out[i] = s > 0.f ? s : expm1f(s);
}
__global__ void fc3_logsoftmax(const float* __restrict__ in, const float* __restrict__ w,
                               const float* __restrict__ bias, float* __restrict__ out, int M) {
  int g = blockIdx.x * blockDim.x + threadIdx.x;
  if (g >= M) return;
  float z0 = bias[0], z1 = bias[1];
  for (int k = 0; k < 64; ++k) {
    float v = in[g * 64 + k];
    z0 += v * w[k * 2 + 0];
    z1 += v * w[k * 2 + 1];
  }
  float m = fmaxf(z0, z1);
  float lse = m + logf(expf(z0 - m) + expf(z1 - m));
  out[g * 2 + 0] = z0 - lse;
  out[g * 2 + 1] = z1 - lse;
}

// ---------- host orchestration ----------
static inline size_t alignup(size_t x) { return (x + 255) & ~(size_t)255; }

extern "C" void kernel_launch(void* const* d_in, const int* in_sizes, int n_in,
                              void* d_out, int out_size, void* d_ws, size_t ws_size,
                              hipStream_t stream) {
  (void)in_sizes; (void)n_in; (void)out_size; (void)ws_size;
  const float* x    = (const float*)d_in[0];
  const float* eatt = (const float*)d_in[1];
  const int*   ei   = (const int*)d_in[2];
  const int*   batch = (const int*)d_in[3];
  const float *Wl[3], *Wr[3], *We[3], *att[3], *bias[3], *gam[3], *bet[3];
  for (int i = 0; i < 3; ++i) {
    int base = 4 + i * 7;
    Wl[i]  = (const float*)d_in[base + 0];
    Wr[i]  = (const float*)d_in[base + 1];
    We[i]  = (const float*)d_in[base + 2];
    att[i] = (const float*)d_in[base + 3];
    bias[i]= (const float*)d_in[base + 4];
    gam[i] = (const float*)d_in[base + 5];
    bet[i] = (const float*)d_in[base + 6];
  }
  const float* fc1_w = (const float*)d_in[25];
  const float* fc1_b = (const float*)d_in[26];
  const float* fc2_w = (const float*)d_in[27];
  const float* fc2_b = (const float*)d_in[28];
  const float* fc3_w = (const float*)d_in[29];
  const float* fc3_b = (const float*)d_in[30];

  char* p = (char*)d_ws;
  auto take = [&](size_t bytes) { void* r = (void*)p; p += alignup(bytes); return r; };
  float*          xl     = (float*)take((size_t)NP * DD * 4);   // padded rows for WMMA
  float*          xr     = (float*)take((size_t)NP * DD * 4);
  float*          outb   = (float*)take((size_t)NN * DD * 4);
  float*          hbuf   = (float*)take((size_t)NN * DD * 4);
  unsigned short* hbf    = (unsigned short*)take((size_t)NP * DD * 2);
  unsigned short* wlbf   = (unsigned short*)take((size_t)DD * DD * 2);
  unsigned short* wrbf   = (unsigned short*)take((size_t)DD * DD * 2);
  float*          logit  = (float*)take((size_t)EE * HH * 4);
  unsigned*       nmax   = (unsigned*)take((size_t)NN * HH * 4);
  float*          den    = (float*)take((size_t)NN * HH * 4);
  float*          pooled = (float*)take((size_t)GG * DD * 4);
  float*          counts = (float*)take((size_t)GG * 4);
  float*          z1     = (float*)take((size_t)GG * 128 * 4);
  float*          z2     = (float*)take((size_t)GG * 64 * 4);

  const int B = 256;
  const int nd_blk  = (NN * DD + B - 1) / B;       // 40000
  const int nh_blk  = (NN * HH + B - 1) / B;
  const int eh_blk  = (EE * HH + B - 1) / B;       // per-thread edge kernels
  const int ehw_blk = (EE * HH) / (B / 32);        // per-wave edge kernels: 160000
  const int dd2_blk = (DD * DD + B - 1) / B;
  const int pad_w   = (NP - NN) * DD / 2;          // pad words of hbf (u32)
  // WMMA GEMM: exact grid, 64x16 strip per wave: NP/64 strips x DD/16 N-tiles
  const int gemm_blk = (NP / 64) * (DD / 16) * 32 / B;   // 1256

  const int* src = ei;
  const int* dst = ei + EE;

  for (int L = 0; L < 3; ++L) {
    // --- node transforms xl = h@Wl, xr = h@Wr ---
    if (L == 0) {
      gemm_f32_small<<<nd_blk, B, 0, stream>>>(x, Wl[0], xl, NN, NFD, DD);
      gemm_f32_small<<<nd_blk, B, 0, stream>>>(x, Wr[0], xr, NN, NFD, DD);
    } else {
      cvt_bf16<<<nd_blk, B, 0, stream>>>(hbf, hbuf, NN * DD);
      fill_u32<<<(pad_w + B - 1) / B, B, 0, stream>>>((unsigned*)(hbf + (size_t)NN * DD), 0u, pad_w);
      pack_b_bf16<<<dd2_blk, B, 0, stream>>>(wlbf, Wl[L]);
      pack_b_bf16<<<dd2_blk, B, 0, stream>>>(wrbf, Wr[L]);
      gemm_bf16_wmma<<<gemm_blk, B, 0, stream>>>(hbf, wlbf, xl);
      gemm_bf16_wmma<<<gemm_blk, B, 0, stream>>>(hbf, wrbf, xr);
    }
    // --- attention + segment softmax + scatter ---
    fill_u32<<<nh_blk, B, 0, stream>>>(nmax, FMAP_NEG_INF, NN * HH);
    fill_f32<<<nh_blk, B, 0, stream>>>(den, 0.f, NN * HH);
    fill_f32<<<nd_blk, B, 0, stream>>>(outb, 0.f, NN * DD);
    edge_logit<<<ehw_blk, B, 0, stream>>>(xl, xr, eatt, We[L], att[L], src, dst, logit, nmax);
    edge_exp<<<eh_blk, B, 0, stream>>>(logit, nmax, den, dst);
    edge_aggr<<<ehw_blk, B, 0, stream>>>(xl, logit, den, src, dst, outb);
    post_bn_elu<<<nd_blk, B, 0, stream>>>(outb, bias[L], gam[L], bet[L], hbuf);
  }

  // --- global mean pool + MLP head ---
  fill_f32<<<(GG * DD + B - 1) / B, B, 0, stream>>>(pooled, 0.f, GG * DD);
  fill_f32<<<1, B, 0, stream>>>(counts, 0.f, GG);
  pool_count<<<(NN + B - 1) / B, B, 0, stream>>>(batch, counts);
  pool_sum<<<nd_blk, B, 0, stream>>>(hbuf, batch, pooled);
  pool_div<<<(GG * DD + B - 1) / B, B, 0, stream>>>(pooled, counts);
  fc_elu<<<(GG * 128 + B - 1) / B, B, 0, stream>>>(pooled, fc1_w, fc1_b, z1, GG, DD, 128);
  fc_elu<<<(GG * 64 + B - 1) / B, B, 0, stream>>>(z1, fc2_w, fc2_b, z2, GG, 128, 64);
  fc3_logsoftmax<<<1, 128, 0, stream>>>(z2, fc3_w, fc3_b, (float*)d_out, GG);
}
